// BidirectionalMambaBlock_8710193676776
// MI455X (gfx1250) — matrix operand order repaired and checked by reading, not compiled, not run on hardware
//
#include <hip/hip_runtime.h>

typedef __bf16 v16bf __attribute__((ext_vector_type(16)));
typedef __bf16 v8bf  __attribute__((ext_vector_type(8)));
typedef __bf16 v4bf  __attribute__((ext_vector_type(4)));
typedef float  v8f   __attribute__((ext_vector_type(8)));
typedef float  v4f   __attribute__((ext_vector_type(4)));
typedef unsigned int u32x4 __attribute__((ext_vector_type(4)));
typedef unsigned int u32x8 __attribute__((ext_vector_type(8)));

#define B_     2
#define L_     1024
#define DMODEL 768
#define DINNER 1536
#define DCONV  4
#define DSTATE 16
#define DTRANK 48
#define XDW    (DTRANK + 2 * DSTATE)   /* 80 */
#define ML     (B_ * L_)               /* 2048 */

__device__ __forceinline__ float silu_f(float x)     { return x / (1.f + __expf(-x)); }
__device__ __forceinline__ float softplus_f(float x) { return (x > 20.f) ? x : log1pf(__expf(x)); }

// ---------------------------------------------------------------------------
// f32 (KxN) -> bf16 transposed (NxK) so GEMM B tiles are K-contiguous.
// ---------------------------------------------------------------------------
__global__ void cvtT_bf16_kernel(const float* __restrict__ in, __bf16* __restrict__ out,
                                 int K, int N) {
  int i = blockIdx.x * blockDim.x + threadIdx.x;
  if (i >= K * N) return;
  int n = i / K, k = i % K;
  out[(size_t)n * K + k] = (__bf16)in[(size_t)k * N + n];
}

// ---------------------------------------------------------------------------
// WMMA GEMM:  C = A(MxK, f32, lda) * B (given as B^T: NxK bf16, ldbt)
//   mode 0: C0 = v     mode 1: split -> C0 | silu -> C1     mode 2: softplus(v+bias)
// Block: 256 threads = 8 waves; tile 32(M) x 128(N), K-step 64.
// A tile: VALU f32->bf16 conversion into LDS (all waves).
// B tile: Tensor Data Mover DMA into LDS (wave 0 issues D#; TENSORcnt wait).
//         D# pad fields reproduce the padded [128][72] LDS layout; tensor-dim
//         OOB zero-fills the K/N tails (those GEMMs start at k0=n0=0).
// ---------------------------------------------------------------------------
__global__ __launch_bounds__(256) void gemm_wmma_bf16(
    const float* __restrict__ A, const __bf16* __restrict__ Bt,
    float* __restrict__ C0, float* __restrict__ C1, const float* __restrict__ bias,
    int M, int N, int K, int lda, int ldbt, int ld0, int ld1, int split, int mode)
{
  __shared__ __align__(16) __bf16 Als[32][72];    // 32 rows x 64 k (pad 8)
  __shared__ __align__(16) __bf16 Bls[128][72];   // 128 cols(N-major) x 64 k (pad 8)

  const int tid  = threadIdx.x;
  const int wave = tid >> 5;
  const int lane = tid & 31;
  const int wm   = wave >> 2;           // 0..1
  const int wn   = wave & 3;            // 0..3
  const int m0   = blockIdx.y * 32;
  const int n0   = blockIdx.x * 128;
  const int half = (lane < 16) ? 0 : 1;
  const int l15  = lane & 15;

  v8f acc0 = {}, acc1 = {};

  const int ra  = l15 + wm * 16;        // A row within tile
  const int ka  = half * 16;            // A k-phase
  const int cb0 = l15 + wn * 32;        // B col (first 16x16)
  const int kb  = half * 16;            // B k-phase
  const unsigned ldsB = (unsigned)(size_t)(&Bls[0][0]);  // LDS byte offset of B tile

  for (int k0 = 0; k0 < K; k0 += 64) {
    // ---- B tile via TDM (one descriptor, wave 0) --------------------------
    if (wave == 0) {
      unsigned long long ga =
          (unsigned long long)(size_t)(Bt + (size_t)n0 * ldbt + k0);
      u32x4 g0;
      g0[0] = 1u;                                   // count=1, user mode
      g0[1] = ldsB;                                 // lds_addr (bytes)
      g0[2] = (unsigned)ga;                         // global_addr[31:0]
      g0[3] = ((unsigned)(ga >> 32) & 0x01FFFFFFu)  // global_addr[56:32]
            | (2u << 30);                           // type=2 ("image")
      u32x8 g1;
      g1[0] = (1u << 16)        // data_size = 1 -> 2 bytes (bf16)
            | (1u << 20)        // pad_enable
            | (4u << 22)        // pad_interval = 32 DWORDs (one 64-elem row)
            | (3u << 25);       // pad_amount   = 4 DWORDs (8-elem row pad)
      g1[1] = ((unsigned)K & 0xFFFFu) << 16;                          // dim0[15:0]
      g1[2] = (((unsigned)K >> 16) & 0xFFFFu)                         // dim0[31:16]
            | (((unsigned)N & 0xFFFFu) << 16);                        // dim1[15:0]
      g1[3] = (((unsigned)N >> 16) & 0xFFFFu)                         // dim1[31:16]
            | (64u << 16);                                            // tile_dim0
      g1[4] = 128u;                                                   // tile_dim1
      g1[5] = (unsigned)ldbt;                                         // dim0_stride
      g1[6] = 0u;
      g1[7] = 0u;
      asm volatile("tensor_load_to_lds %0, %1" :: "s"(g0), "s"(g1) : "memory");
    }

    // ---- A tile: f32 -> bf16 staging (all waves) --------------------------
    const bool fastA = (m0 + 32 <= M) && (k0 + 64 <= K);
    if (fastA) {
      #pragma unroll
      for (int c = tid; c < 512; c += 256) {
        int r = c >> 4, coff = (c & 15) * 4;
        v4f fv = *(const v4f*)&A[(size_t)(m0 + r) * lda + (k0 + coff)];
        v4bf bv;
        #pragma unroll
        for (int j = 0; j < 4; ++j) bv[j] = (__bf16)fv[j];
        *(v4bf*)&Als[r][coff] = bv;
      }
    } else {
      for (int i = tid; i < 32 * 64; i += 256) {
        int r = i >> 6, c = i & 63;
        int mg = m0 + r, kg = k0 + c;
        int mc = mg < M ? mg : M - 1;
        int kc = kg < K ? kg : K - 1;
        float v = A[(size_t)mc * lda + kc];           // clamped addr, no branch
        Als[r][c] = (__bf16)((mg < M && kg < K) ? v : 0.f);
      }
    }

    if (wave == 0) __builtin_amdgcn_s_wait_tensorcnt(0);
    __syncthreads();

    #pragma unroll
    for (int kk = 0; kk < 64; kk += 32) {
      v8bf a0 = *(const v8bf*)&Als[ra][kk + ka];
      v8bf a1 = *(const v8bf*)&Als[ra][kk + ka + 8];
      v16bf af = __builtin_shufflevector(a0, a1, 0,1,2,3,4,5,6,7,8,9,10,11,12,13,14,15);
      v8bf b0 = *(const v8bf*)&Bls[cb0][kk + kb];
      v8bf b1 = *(const v8bf*)&Bls[cb0][kk + kb + 8];
      v16bf bf0 = __builtin_shufflevector(b0, b1, 0,1,2,3,4,5,6,7,8,9,10,11,12,13,14,15);
      v8bf b2 = *(const v8bf*)&Bls[cb0 + 16][kk + kb];
      v8bf b3 = *(const v8bf*)&Bls[cb0 + 16][kk + kb + 8];
      v16bf bf1 = __builtin_shufflevector(b2, b3, 0,1,2,3,4,5,6,7,8,9,10,11,12,13,14,15);

      acc0 = __builtin_amdgcn_wmma_f32_16x16x32_bf16(false, af, false, bf0,
                                                     (short)0, acc0, false, false);
      acc1 = __builtin_amdgcn_wmma_f32_16x16x32_bf16(false, af, false, bf1,
                                                     (short)0, acc1, false, false);
    }
    __syncthreads();
  }

  // Epilogue: acc reg r -> row (r + 8*half), col = lane&15 in each 16x16 tile
  const int rbase = wm * 16 + half * 8;
  const int cg0   = n0 + wn * 32 + l15;
  #pragma unroll
  for (int t = 0; t < 2; ++t) {
    int cg = cg0 + t * 16;
    if (cg >= N) continue;
    #pragma unroll
    for (int r = 0; r < 8; ++r) {
      int mg = m0 + rbase + r;
      if (mg >= M) continue;
      float v = (t == 0) ? acc0[r] : acc1[r];
      if (mode == 0) {
        C0[(size_t)mg * ld0 + cg] = v;
      } else if (mode == 1) {
        if (cg < split) C0[(size_t)mg * ld0 + cg] = v;
        else            C1[(size_t)mg * ld1 + (cg - split)] = silu_f(v);
      } else {
        C0[(size_t)mg * ld0 + cg] = softplus_f(v + bias[cg]);
      }
    }
  }
}

// ---------------------------------------------------------------------------
// Causal depthwise conv (taps=4) + bias + SiLU
// ---------------------------------------------------------------------------
__global__ void dwconv_silu_kernel(const float* __restrict__ x, const float* __restrict__ w,
                                   const float* __restrict__ b, float* __restrict__ y, int total) {
  int i = blockIdx.x * blockDim.x + threadIdx.x;
  if (i >= total) return;
  int d  = i % DINNER;
  int l  = (i / DINNER) % L_;
  int bb = i / (DINNER * L_);
  float acc = b[d];
  #pragma unroll
  for (int t = 0; t < DCONV; ++t) {
    int j = l - (DCONV - 1) + t;
    if (j >= 0) acc += w[d * DCONV + t] * x[((size_t)bb * L_ + j) * DINNER + d];
  }
  y[i] = silu_f(acc);
}

// ---------------------------------------------------------------------------
// Time gather: y[b,l,:] = x[b, ids[b,l], :]
// ---------------------------------------------------------------------------
__global__ void gather_time_kernel(const float* __restrict__ x, const int* __restrict__ ids,
                                   float* __restrict__ y, int total) {
  int i = blockIdx.x * blockDim.x + threadIdx.x;
  if (i >= total) return;
  int d  = i % DINNER;
  int l  = (i / DINNER) % L_;
  int bb = i / (DINNER * L_);
  int j  = ids[bb * L_ + l];
  y[i] = x[((size_t)bb * L_ + j) * DINNER + d];
}

// ---------------------------------------------------------------------------
// Selective scan: one 16-lane half-wave per (b,d); lane = state index n.
// ---------------------------------------------------------------------------
__global__ __launch_bounds__(256) void scan_kernel(
    const float* __restrict__ u, const float* __restrict__ delta,
    const float* __restrict__ xd, const float* __restrict__ A_log,
    const float* __restrict__ Dv, float* __restrict__ y)
{
  int g = blockIdx.x * (blockDim.x >> 4) + (threadIdx.x >> 4);
  int n = threadIdx.x & 15;
  if (g >= B_ * DINNER) return;
  int b = g / DINNER, d = g % DINNER;
  float a  = -__expf(A_log[d * DSTATE + n]);
  float Dd = Dv[d];
  float h  = 0.f;
  for (int l = 0; l < L_; ++l) {
    size_t base = (size_t)b * L_ + l;
    float dlt = delta[base * DINNER + d];
    float uu  = u[base * DINNER + d];
    float bm  = xd[base * XDW + DTRANK + n];
    float cm  = xd[base * XDW + DTRANK + DSTATE + n];
    h = __expf(dlt * a) * h + dlt * bm * uu;
    float c = h * cm;
    #pragma unroll
    for (int off = 8; off; off >>= 1) c += __shfl_xor(c, off, 16);
    if (n == 0) y[base * DINNER + d] = c + uu * Dd;
  }
}

// ---------------------------------------------------------------------------
// Combine: y = y_fwd*res + gather(y_bwd*res)
// ---------------------------------------------------------------------------
__global__ void combine_kernel(const float* __restrict__ yf, const float* __restrict__ yb,
                               const float* __restrict__ res, const int* __restrict__ ids,
                               float* __restrict__ y, int total) {
  int i = blockIdx.x * blockDim.x + threadIdx.x;
  if (i >= total) return;
  int d  = i % DINNER;
  int l  = (i / DINNER) % L_;
  int bb = i / (DINNER * L_);
  int j  = ids[bb * L_ + l];
  size_t pj = ((size_t)bb * L_ + j) * DINNER + d;
  y[i] = yf[i] * res[i] + yb[pj] * res[pj];
}

// ---------------------------------------------------------------------------
extern "C" void kernel_launch(void* const* d_in, const int* in_sizes, int n_in,
                              void* d_out, int out_size, void* d_ws, size_t ws_size,
                              hipStream_t stream)
{
  const float* x       = (const float*)d_in[0];
  const int*   ids     = (const int*)  d_in[1];
  const float* W_in    = (const float*)d_in[2];
  const float* W_out   = (const float*)d_in[3];
  const float* conv_w  = (const float*)d_in[4];
  const float* conv_b  = (const float*)d_in[5];
  const float* conv_wf = (const float*)d_in[6];
  const float* conv_bf = (const float*)d_in[7];
  const float* W_xp    = (const float*)d_in[8];
  const float* W_xpf   = (const float*)d_in[9];
  const float* W_dt    = (const float*)d_in[10];
  const float* b_dt    = (const float*)d_in[11];
  const float* W_dtf   = (const float*)d_in[12];
  const float* b_dtf   = (const float*)d_in[13];
  const float* A_log   = (const float*)d_in[14];
  const float* Dv      = (const float*)d_in[15];
  const float* A_logf  = (const float*)d_in[16];
  const float* Dvf     = (const float*)d_in[17];

  const size_t BLD = (size_t)B_ * L_ * DINNER;
  const size_t BLX = (size_t)B_ * L_ * XDW;

  float* fb  = (float*)d_ws;
  float* xs  = fb + 0 * BLD;   // xs -> xflip -> y_ssm_bwd
  float* res = fb + 1 * BLD;
  float* xc  = fb + 2 * BLD;   // conv-fwd output -> final combined y
  float* xcf = fb + 3 * BLD;
  float* dlt = fb + 4 * BLD;   // delta fwd, then delta bwd
  float* ysf = fb + 5 * BLD;
  float* xdf = fb + 6 * BLD;   // (B,L,80)
  float* xdb = xdf + BLX;

  __bf16* bfp     = (__bf16*)(xdb + BLX);
  const size_t nWin  = (size_t)DMODEL * 2 * DINNER;
  const size_t nWout = (size_t)DINNER * DMODEL;
  const size_t nWxp  = (size_t)DINNER * XDW;
  const size_t nWdt  = (size_t)DTRANK * DINNER;
  __bf16* WinT  = bfp;             // (3072 x 768)
  __bf16* WoutT = WinT  + nWin;    // (768 x 1536)
  __bf16* WxpT  = WoutT + nWout;   // (80 x 1536)
  __bf16* WxpfT = WxpT  + nWxp;
  __bf16* WdtT  = WxpfT + nWxp;    // (1536 x 48)
  __bf16* WdtfT = WdtT  + nWdt;

  auto cvtT = [&](const float* s, __bf16* d, int K, int N) {
    int n = K * N;
    cvtT_bf16_kernel<<<(n + 255) / 256, 256, 0, stream>>>(s, d, K, N);
  };
  cvtT(W_in,  WinT,  DMODEL, 2 * DINNER);
  cvtT(W_out, WoutT, DINNER, DMODEL);
  cvtT(W_xp,  WxpT,  DINNER, XDW);
  cvtT(W_xpf, WxpfT, DINNER, XDW);
  cvtT(W_dt,  WdtT,  DTRANK, DINNER);
  cvtT(W_dtf, WdtfT, DTRANK, DINNER);

  auto gemm = [&](const float* A, const __bf16* Bt, float* C0, float* C1, const float* bias,
                  int M, int N, int K, int lda, int ld0, int ld1, int split, int mode) {
    dim3 grid((N + 127) / 128, (M + 31) / 32);
    gemm_wmma_bf16<<<grid, 256, 0, stream>>>(A, Bt, C0, C1, bias,
                                             M, N, K, lda, /*ldbt=*/K, ld0, ld1, split, mode);
  };

  const int totalBLD = (int)BLD;
  const int eg = (totalBLD + 255) / 256;
  const int scanBlocks = (B_ * DINNER + 15) / 16;

  // 1) xr = x @ W_in : split -> xs | silu -> res
  gemm(x, WinT, xs, res, nullptr, ML, 2 * DINNER, DMODEL, DMODEL, DINNER, DINNER, DINNER, 1);
  // 2) xc = silu(causal_dwconv(xs))
  dwconv_silu_kernel<<<eg, 256, 0, stream>>>(xs, conv_w, conv_b, xc, totalBLD);
  // 3) xd_fwd = xc @ W_xproj
  gemm(xc, WxpT, xdf, nullptr, nullptr, ML, XDW, DINNER, DINNER, XDW, 0, 0, 0);
  // 4) delta_fwd = softplus(xd_fwd[:, :48] @ W_dt + b_dt)
  gemm(xdf, WdtT, dlt, nullptr, b_dt, ML, DINNER, DTRANK, XDW, DINNER, 0, 0, 2);
  // 5) forward selective scan
  scan_kernel<<<scanBlocks, 256, 0, stream>>>(xc, dlt, xdf, A_log, Dv, ysf);
  // 6) xflip = gather(xc, ids)  (xs slot)
  gather_time_kernel<<<eg, 256, 0, stream>>>(xc, ids, xs, totalBLD);
  // 7) xcf = silu(causal_dwconv(xflip))
  dwconv_silu_kernel<<<eg, 256, 0, stream>>>(xs, conv_wf, conv_bf, xcf, totalBLD);
  // 8) xd_bwd = xcf @ W_xproj_f
  gemm(xcf, WxpfT, xdb, nullptr, nullptr, ML, XDW, DINNER, DINNER, XDW, 0, 0, 0);
  // 9) delta_bwd
  gemm(xdb, WdtfT, dlt, nullptr, b_dtf, ML, DINNER, DTRANK, XDW, DINNER, 0, 0, 2);
  // 10) backward selective scan -> xs slot
  scan_kernel<<<scanBlocks, 256, 0, stream>>>(xcf, dlt, xdb, A_logf, Dvf, xs);
  // 11) y = ysf*res + gather(y_bwd*res) -> xc slot
  combine_kernel<<<eg, 256, 0, stream>>>(ysf, xs, res, ids, xc, totalBLD);
  // 12) out = y @ W_out
  gemm(xc, WoutT, (float*)d_out, nullptr, nullptr, ML, DMODEL, DINNER, DINNER, DMODEL, 0, 0, 0);
}
